// NetworkLayer_42975442764619
// MI455X (gfx1250) — matrix operand order repaired and checked by reading, not compiled, not run on hardware
//
#include <hip/hip_runtime.h>

typedef __attribute__((ext_vector_type(16))) _Float16 v16h;
typedef __attribute__((ext_vector_type(8)))  _Float16 v8h;
typedef __attribute__((ext_vector_type(8)))  float    v8f;

#define DEVINL __device__ __forceinline__

// Padded LDS strides (in halves): row-stride mod 64 words is a non-trivial
// residue -> each wave-wide ds_load_b128 spreads over the 64 banks with the
// optimal 2-way occupancy (vs 16-way conflicts at stride 256).
#define SP 168   // scalars stride: 336 B = 84 words, 84 % 64 = 20
#define HP 264   // hidden  stride: 528 B = 132 words, 132 % 64 = 4

// ---------------------------------------------------------------------------
// Pack f32 weight W[K][256] (row-major) into WMMA-B fragment-ready layout.
// 16x16x32 f16 WMMA B (32x16 K-major): lane l holds column n = l%16;
// halves h=0..7 hold K = (l/16)*8 + h, h=8..15 hold K = 16 + (l/16)*8 + (h-8).
// Packed: Wp[(tk*16+tn)*512 + lane*16 + h], zero-padded for k >= Kreal.
// ---------------------------------------------------------------------------
__global__ void pack_weights(const float* __restrict__ W, _Float16* __restrict__ Wp,
                             int Kreal, int KT) {
  int total = KT * 16 * 512;
  for (int idx = blockIdx.x * blockDim.x + threadIdx.x; idx < total;
       idx += gridDim.x * blockDim.x) {
    int h    = idx & 15;
    int lane = (idx >> 4) & 31;
    int tile = idx >> 9;            // tk*16 + tn
    int tk = tile >> 4, tn = tile & 15;
    int k = tk * 32 + (h >> 3) * 16 + (lane >> 4) * 8 + (h & 7);
    int n = tn * 16 + (lane & 15);
    float v = (k < Kreal) ? W[k * 256 + n] : 0.0f;
    Wp[idx] = (_Float16)v;
  }
}

DEVINL v16h load_afrag(const _Float16* ar) {
  v8h lo = *(const v8h*)(ar);
  v8h hi = *(const v8h*)(ar + 16);
  return __builtin_shufflevector(lo, hi, 0, 1, 2, 3, 4, 5, 6, 7,
                                         8, 9, 10, 11, 12, 13, 14, 15);
}

// ---------------------------------------------------------------------------
// One layer: Out[128][256] = act(A[128][K] @ W + bias), A/Out in LDS.
// Each of the 16 waves owns ONE N-tile column (n = w*16..+15) and all 8
// M-tiles: per k-step one B fragment (weights hit L2 once per block,
// double-buffered across k-steps) and 8 A fragments from LDS.
// A fragments are double-buffered across M-tiles so the next fragment's
// ds_load_b128 pair is in flight while the current WMMA executes
// (wait becomes dscnt<=2 instead of dscnt==0).
// A frag (16x32 f16): lane l row m = l%16; halves 0..7 = K lh*8, 8..15 =
// K 16+lh*8. D (16x16 f32): lane l -> n = l%16, m = r + 8*lh.
// ---------------------------------------------------------------------------
DEVINL void gemm_layer(const _Float16* __restrict__ A, int lda, int KT,
                       const _Float16* __restrict__ Wp, const float* __restrict__ bias,
                       _Float16* __restrict__ Out, int ldo, bool relu) {
  const int tid  = threadIdx.x;
  const int lane = tid & 31;
  const int w    = tid >> 5;      // N-tile 0..15
  const int lr   = lane & 15;
  const int lh   = lane >> 4;

  const int n = w * 16 + lr;
  const float bv = bias[n];                       // invariant across M-tiles
  const _Float16* wcol = Wp + w * 512 + lane * 16;

  v8f acc[8] = {};
  v16h bcur = *(const v16h*)(wcol);               // B frag for tk = 0

  for (int tk = 0; tk < KT; ++tk) {
    v16h bnext = (tk + 1 < KT) ? *(const v16h*)(wcol + (tk + 1) * 16 * 512)
                               : bcur;            // prefetch next k-step
    const _Float16* abase = A + lr * lda + tk * 32 + lh * 8;
    v16h acur = load_afrag(abase);                // M-tile 0
#pragma unroll
    for (int mi = 0; mi < 8; ++mi) {
      v16h anext = (mi + 1 < 8) ? load_afrag(abase + (mi + 1) * 16 * lda)
                                : acur;           // prefetch next M-tile
      acc[mi] = __builtin_amdgcn_wmma_f32_16x16x32_f16(
          false, acur, false, bcur, (short)0, acc[mi], false, false);
      acur = anext;
    }
    bcur = bnext;
  }

#pragma unroll
  for (int mi = 0; mi < 8; ++mi) {
#pragma unroll
    for (int r = 0; r < 8; ++r) {
      int m = mi * 16 + lh * 8 + r;
      float v = acc[mi][r] + bv;
      if (relu) v = (v > 0.0f) ? v : 0.01f * v;
      Out[m * ldo + n] = (_Float16)v;
    }
  }
}

// ---------------------------------------------------------------------------
// Fully fused per-batch kernel. Block = 512 threads (16 wave32), grid = B.
// Dynamic LDS: header (~4.7 KB) + two padded f16 activation buffers (66 KB).
//   scalars -> bufB (128 x SP); L0: bufB -> bufA; L1: bufA -> bufB;
//   L2: bufB -> bufA (fk); final: out[b][o][d] = sum_i fk[i][o]*x[i][d]/128
// ---------------------------------------------------------------------------
__global__ __launch_bounds__(512)
void fused_mlp(const float* __restrict__ x, const float* __restrict__ u,
               const float* __restrict__ basis,
               const _Float16* __restrict__ W0p, const float* __restrict__ b0,
               const _Float16* __restrict__ W1p, const float* __restrict__ b1,
               const _Float16* __restrict__ W2p, const float* __restrict__ b2,
               float* __restrict__ out) {
  extern __shared__ char smem[];
  float* xs   = (float*)smem;                     // 128*3 floats
  float* bas  = xs + 384;                         // 9 floats
  float* us   = bas + 12;                         // 2 floats (+pad to 400)
  float* red  = us + 4;                           // 256*3 floats partial sums
  _Float16* bufA = (_Float16*)(smem + 4800);      // 128 x HP halves
  _Float16* bufB = bufA + 128 * HP;               // 128 x HP halves

  const int b   = blockIdx.x;
  const int tid = threadIdx.x;

  for (int t = tid; t < 384; t += 512) xs[t] = x[b * 384 + t];
  if (tid < 9) bas[tid] = basis[b * 9 + tid];
  if (tid < 2) us[tid]  = u[b * 2 + tid];
  __syncthreads();

  // ---- scalars: [u(2) | norm(1) | basis_proj(3) | dots(128) | pad] ----
  _Float16* S = bufB;                             // 128 x SP
  for (int idx = tid; idx < 128 * SP; idx += 512) {
    int i = idx / SP, k = idx - i * SP;
    float x0 = xs[i * 3 + 0], x1 = xs[i * 3 + 1], x2 = xs[i * 3 + 2];
    float v;
    if (k < 2) {
      v = us[k];
    } else if (k < 6) {
      float nrm = sqrtf(x0 * x0 + x1 * x1 + x2 * x2);
      if (k == 2) {
        v = nrm;
      } else {
        int j = k - 3;
        v = (x0 * bas[j * 3] + x1 * bas[j * 3 + 1] + x2 * bas[j * 3 + 2]) / nrm;
      }
    } else if (k < 134) {
      int j = k - 6;
      v = x0 * xs[j * 3] + x1 * xs[j * 3 + 1] + x2 * xs[j * 3 + 2];
    } else {
      v = 0.0f;                                   // zero-pad K to 160
    }
    S[idx] = (_Float16)v;
  }
  __syncthreads();

  gemm_layer(S,    SP, 5, W0p, b0, bufA, HP, true);   // h0 -> bufA
  __syncthreads();
  gemm_layer(bufA, HP, 8, W1p, b1, bufB, HP, true);   // h1 -> bufB
  __syncthreads();
  gemm_layer(bufB, HP, 8, W2p, b2, bufA, HP, false);  // fk -> bufA
  __syncthreads();

  // ---- out[b][o][d] = (1/128) * sum_i fk[i][o] * x[i][d] ----
  // All 512 threads: thread (o, half) sums 64 rows; halves combined via LDS.
  {
    const int o  = tid & 255;
    const int hf = tid >> 8;                      // 0 or 1
    const int i0 = hf * 64;
    float a0 = 0.f, a1 = 0.f, a2 = 0.f;
#pragma unroll 4
    for (int i = i0; i < i0 + 64; ++i) {
      float f = (float)bufA[i * HP + o];
      a0 += f * xs[i * 3 + 0];
      a1 += f * xs[i * 3 + 1];
      a2 += f * xs[i * 3 + 2];
    }
    if (hf == 1) {
      red[o * 3 + 0] = a0; red[o * 3 + 1] = a1; red[o * 3 + 2] = a2;
    }
    __syncthreads();
    if (hf == 0) {
      const float inv = 1.0f / 128.0f;
      out[b * 768 + o * 3 + 0] = (a0 + red[o * 3 + 0]) * inv;
      out[b * 768 + o * 3 + 1] = (a1 + red[o * 3 + 1]) * inv;
      out[b * 768 + o * 3 + 2] = (a2 + red[o * 3 + 2]) * inv;
    }
  }
}

// ---------------------------------------------------------------------------
extern "C" void kernel_launch(void* const* d_in, const int* in_sizes, int n_in,
                              void* d_out, int out_size, void* d_ws, size_t ws_size,
                              hipStream_t stream) {
  const float* x     = (const float*)d_in[0];
  const float* u     = (const float*)d_in[1];
  const float* basis = (const float*)d_in[2];
  const float* W0    = (const float*)d_in[3];
  const float* b0    = (const float*)d_in[4];
  const float* W1    = (const float*)d_in[5];
  const float* b1    = (const float*)d_in[6];
  const float* W2    = (const float*)d_in[7];
  const float* b2    = (const float*)d_in[8];
  float* out = (float*)d_out;

  // Packed weights in workspace: W0p 80KB, W1p/W2p 128KB each (~336KB total).
  _Float16* W0p = (_Float16*)d_ws;
  _Float16* W1p = W0p + 5 * 16 * 512;
  _Float16* W2p = W1p + 8 * 16 * 512;

  pack_weights<<<128, 512, 0, stream>>>(W0, W0p, 134, 5);
  pack_weights<<<128, 512, 0, stream>>>(W1, W1p, 256, 8);
  pack_weights<<<128, 512, 0, stream>>>(W2, W2p, 256, 8);

  const size_t smem = 4800 + (size_t)2 * 128 * HP * sizeof(_Float16); // ~137 KB
  fused_mlp<<<1024, 512, smem, stream>>>(x, u, basis, W0p, b0, W1p, b1, W2p, b2, out);
}
